// GNN_71983651881217
// MI455X (gfx1250) — matrix operand order repaired and checked by reading, not compiled, not run on hardware
//
#include <hip/hip_runtime.h>
#include <hip/hip_bf16.h>

typedef __attribute__((ext_vector_type(16))) _Float16 v16h;
typedef __attribute__((ext_vector_type(8)))  _Float16 v8h;
typedef __attribute__((ext_vector_type(8)))  float    v8f;

#define N_NODES  50000
#define N_EDGES  600000
#define DIM      128
#define N_GRAPHS 64
#define LDS_STRIDE 136   // 128 + 8 halves pad to spread LDS banks

// ---------------------------------------------------------------------------
// Generic u32 fill (zero agg buffer / init segment-max keys)
// ---------------------------------------------------------------------------
__global__ void fill_u32_kernel(unsigned* __restrict__ p, unsigned val, long n) {
    long i = (long)blockIdx.x * blockDim.x + threadIdx.x;
    long stride = (long)gridDim.x * blockDim.x;
    for (; i < n; i += stride) p[i] = val;
}

// ---------------------------------------------------------------------------
// One-shot f32 -> f16 weight conversion (row-major preserved)
// ---------------------------------------------------------------------------
__global__ void f32_to_f16_kernel(const float* __restrict__ src,
                                  _Float16* __restrict__ dst, int n) {
    int i = blockIdx.x * blockDim.x + threadIdx.x;
    if (i < n) dst[i] = (_Float16)src[i];
}

// ---------------------------------------------------------------------------
// Edge scatter-add: agg[dst] += h[src]   (32 threads per edge, float4 each)
// h and agg are both L2-resident (25.6MB each vs 192MB L2).
// ---------------------------------------------------------------------------
__global__ void scatter_add_kernel(const float* __restrict__ h,
                                   const int*   __restrict__ src,
                                   const int*   __restrict__ dst,
                                   float*       __restrict__ agg) {
    long tid = (long)blockIdx.x * blockDim.x + threadIdx.x;
    int  e   = (int)(tid >> 5);
    int  d4  = (int)(tid & 31) * 4;
    if (e >= N_EDGES) return;
    int s = src[e];
    int d = dst[e];
    const float4 v = *(const float4*)(h + (long)s * DIM + d4);
    float* p = agg + (long)d * DIM + d4;
    atomicAdd(p + 0, v.x);
    atomicAdd(p + 1, v.y);
    atomicAdd(p + 2, v.z);
    atomicAdd(p + 3, v.w);
}

// ---------------------------------------------------------------------------
// Fused GraphConv GEMM via WMMA:
//   Out[m,n] = (relu?)( agg[m,:].Wrel[n,:] + h[m,:].Wroot[n,:] + brel[n] )
// Block = one 16-row tile, 8 waves cover the 8 column tiles.
// A tiles (shared by all 8 waves) staged f32->f16 in LDS once per block;
// waves read fragments via aligned 16B LDS loads. B fragments are 16
// contiguous pre-converted f16 per lane -> direct 32B global load.
// ---------------------------------------------------------------------------
__global__ void
__launch_bounds__(256)
gemm_fused_kernel(const float*    __restrict__ A1,     // agg  [N,128] f32
                  const float*    __restrict__ A2,     // h_in [N,128] f32
                  const _Float16* __restrict__ Wrel,   // [128,128] f16 row-major
                  const _Float16* __restrict__ Wroot,  // [128,128] f16
                  const float*    __restrict__ brel,   // [128] f32
                  float*          __restrict__ Out,    // [N,128] f32
                  int relu_on) {
    __shared__ __align__(16) _Float16 As1[16 * LDS_STRIDE];
    __shared__ __align__(16) _Float16 As2[16 * LDS_STRIDE];

    const int t     = threadIdx.x;
    const int rbase = blockIdx.x * 16;

    // ---- Cooperative stage: 16x128 f32 -> f16 into LDS (both A matrices)
    {
        const int row = t >> 4;          // 0..15
        const int col = (t & 15) * 8;    // 0,8,...,120
        const long goff = (long)(rbase + row) * DIM + col;
        float4 p0 = *(const float4*)(A1 + goff);
        float4 p1 = *(const float4*)(A1 + goff + 4);
        v8h h1;
        h1[0] = (_Float16)p0.x; h1[1] = (_Float16)p0.y;
        h1[2] = (_Float16)p0.z; h1[3] = (_Float16)p0.w;
        h1[4] = (_Float16)p1.x; h1[5] = (_Float16)p1.y;
        h1[6] = (_Float16)p1.z; h1[7] = (_Float16)p1.w;
        *(v8h*)&As1[row * LDS_STRIDE + col] = h1;

        float4 q0 = *(const float4*)(A2 + goff);
        float4 q1 = *(const float4*)(A2 + goff + 4);
        v8h h2;
        h2[0] = (_Float16)q0.x; h2[1] = (_Float16)q0.y;
        h2[2] = (_Float16)q0.z; h2[3] = (_Float16)q0.w;
        h2[4] = (_Float16)q1.x; h2[5] = (_Float16)q1.y;
        h2[6] = (_Float16)q1.z; h2[7] = (_Float16)q1.w;
        *(v8h*)&As2[row * LDS_STRIDE + col] = h2;
    }
    __syncthreads();

    const int  lane = t & 31;
    const int  wave = t >> 5;            // column tile 0..7
    const int  m    = lane & 15;
    const bool hi   = lane >= 16;
    const int  ncol = wave * 16 + m;     // B/C/D column (global)

    v8f c = {};
    #pragma unroll
    for (int k0 = 0; k0 < DIM; k0 += 32) {
        // A fragment: element e -> k = k0 + hi*16 + e (matches B)
        //   group1: k = k0 + hi*16 + [0..7]   group2: k = k0 + hi*16 + 8 + [0..7]
        const _Float16* a1p = &As1[m * LDS_STRIDE + k0 + (hi ? 16 : 0)];
        const _Float16* a2p = &As2[m * LDS_STRIDE + k0 + (hi ? 16 : 0)];
        v8h a1lo = *(const v8h*)(a1p);
        v8h a1hi = *(const v8h*)(a1p + 8);
        v8h a2lo = *(const v8h*)(a2p);
        v8h a2hi = *(const v8h*)(a2p + 8);
        v16h a1 = __builtin_shufflevector(a1lo, a1hi,
                   0,1,2,3,4,5,6,7,8,9,10,11,12,13,14,15);
        v16h a2 = __builtin_shufflevector(a2lo, a2hi,
                   0,1,2,3,4,5,6,7,8,9,10,11,12,13,14,15);

        // B fragment: element e -> k = k0 + e + hi*16, col = ncol
        // 16 contiguous f16 -> one aligned 32-byte load
        v16h b1 = *(const v16h*)(Wrel  + (long)ncol * DIM + k0 + (hi ? 16 : 0));
        v16h b2 = *(const v16h*)(Wroot + (long)ncol * DIM + k0 + (hi ? 16 : 0));

        c = __builtin_amdgcn_wmma_f32_16x16x32_f16(false, a1, false, b1,
                                                   (short)0, c, false, false);
        c = __builtin_amdgcn_wmma_f32_16x16x32_f16(false, a2, false, b2,
                                                   (short)0, c, false, false);
    }

    // C/D f32 layout: element r -> row = rbase + r + (hi?8:0), col = ncol
    const float bv = brel[ncol];
    #pragma unroll
    for (int r = 0; r < 8; ++r) {
        float v = c[r] + bv;
        if (relu_on) v = fmaxf(v, 0.0f);
        Out[(long)(rbase + r + (hi ? 8 : 0)) * DIM + ncol] = v;
    }
}

// ---------------------------------------------------------------------------
// Segment max-pool via order-preserving float->u32 mapping + atomicMax(u32)
// ---------------------------------------------------------------------------
__device__ __forceinline__ unsigned f2mono(float f) {
    unsigned u = __float_as_uint(f);
    return (u & 0x80000000u) ? ~u : (u | 0x80000000u);
}
__device__ __forceinline__ float mono2f(unsigned u) {
    unsigned b = (u & 0x80000000u) ? (u ^ 0x80000000u) : ~u;
    return __uint_as_float(b);
}

__global__ void seg_max_kernel(const float* __restrict__ h,
                               const int*   __restrict__ batch,
                               unsigned*    __restrict__ gmono) {
    long tid = (long)blockIdx.x * blockDim.x + threadIdx.x;
    if (tid >= (long)N_NODES * DIM) return;
    int node = (int)(tid >> 7);
    int d    = (int)(tid & 127);
    int g    = batch[node];
    atomicMax(&gmono[(long)g * DIM + d], f2mono(h[tid]));
}

// ---------------------------------------------------------------------------
// Final tiny MLP head: out[g] = relu(g_row.W1^T + b1).W2^T + b2  (64 threads)
// ---------------------------------------------------------------------------
__global__ void mlp_kernel(const unsigned* __restrict__ gmono,
                           const float* __restrict__ W1,  // [5,128]
                           const float* __restrict__ b1,  // [5]
                           const float* __restrict__ W2,  // [1,5]
                           const float* __restrict__ b2,  // [1]
                           float* __restrict__ out) {     // [64]
    int g = threadIdx.x;
    if (g >= N_GRAPHS) return;
    float acc = b2[0];
    #pragma unroll
    for (int j = 0; j < 5; ++j) {
        float s = b1[j];
        for (int d = 0; d < DIM; ++d)
            s += mono2f(gmono[(long)g * DIM + d]) * W1[j * DIM + d];
        acc += fmaxf(s, 0.0f) * W2[j];
    }
    out[g] = acc;
}

// ---------------------------------------------------------------------------
// Launch: weight f16 convert, 3 x (zero, scatter, WMMA GEMM), pool, MLP
// ---------------------------------------------------------------------------
extern "C" void kernel_launch(void* const* d_in, const int* in_sizes, int n_in,
                              void* d_out, int out_size, void* d_ws, size_t ws_size,
                              hipStream_t stream) {
    const float* x       = (const float*)d_in[0];
    const int*   eidx    = (const int*)d_in[1];   // [2, N_EDGES]
    const int*   batch   = (const int*)d_in[2];
    const float* Wf[6]   = { (const float*)d_in[3],  (const float*)d_in[5],
                             (const float*)d_in[6],  (const float*)d_in[8],
                             (const float*)d_in[9],  (const float*)d_in[11] };
    const float* b_rel1  = (const float*)d_in[4];
    const float* b_rel2  = (const float*)d_in[7];
    const float* b_rel3  = (const float*)d_in[10];
    const float* W1      = (const float*)d_in[12];
    const float* b1      = (const float*)d_in[13];
    const float* W2      = (const float*)d_in[14];
    const float* b2      = (const float*)d_in[15];
    float* out = (float*)d_out;

    const int* src = eidx;            // edge_index[0,:]
    const int* dst = eidx + N_EDGES;  // edge_index[1,:]

    // Workspace layout
    const long nd   = (long)N_NODES * DIM;          // 6.4M floats
    float*    agg   = (float*)d_ws;
    float*    hA    = agg + nd;
    float*    hB    = hA + nd;
    unsigned* gmono = (unsigned*)(hB + nd);
    _Float16* w16   = (_Float16*)(gmono + (long)N_GRAPHS * DIM);
    const int wmat  = DIM * DIM;                    // 16384
    _Float16* Wrel1h  = w16 + 0 * wmat;
    _Float16* Wroot1h = w16 + 1 * wmat;
    _Float16* Wrel2h  = w16 + 2 * wmat;
    _Float16* Wroot2h = w16 + 3 * wmat;
    _Float16* Wrel3h  = w16 + 4 * wmat;
    _Float16* Wroot3h = w16 + 5 * wmat;
    _Float16* Wh[6] = { Wrel1h, Wroot1h, Wrel2h, Wroot2h, Wrel3h, Wroot3h };

    const dim3 blk(256);
    const int  fillGrid    = 2048;
    const int  scatterGrid = (int)(((long)N_EDGES * 32 + 255) / 256);  // 75000
    const int  gemmGrid    = N_NODES / 16;                             // 3125
    const int  segGrid     = (int)((nd + 255) / 256);                  // 25000

    // ---- Pre-convert all 6 weight matrices to f16 (once per launch)
    for (int i = 0; i < 6; ++i)
        f32_to_f16_kernel<<<(wmat + 255) / 256, blk, 0, stream>>>(Wf[i], Wh[i], wmat);

    // ---- Layer 1: x -> hA (ReLU)
    fill_u32_kernel<<<fillGrid, blk, 0, stream>>>((unsigned*)agg, 0u, nd);
    scatter_add_kernel<<<scatterGrid, blk, 0, stream>>>(x, src, dst, agg);
    gemm_fused_kernel<<<gemmGrid, blk, 0, stream>>>(agg, x, Wrel1h, Wroot1h,
                                                    b_rel1, hA, 1);
    // ---- Layer 2: hA -> hB (ReLU)
    fill_u32_kernel<<<fillGrid, blk, 0, stream>>>((unsigned*)agg, 0u, nd);
    scatter_add_kernel<<<scatterGrid, blk, 0, stream>>>(hA, src, dst, agg);
    gemm_fused_kernel<<<gemmGrid, blk, 0, stream>>>(agg, hA, Wrel2h, Wroot2h,
                                                    b_rel2, hB, 1);
    // ---- Layer 3: hB -> hA (no ReLU; hA reused as h3)
    fill_u32_kernel<<<fillGrid, blk, 0, stream>>>((unsigned*)agg, 0u, nd);
    scatter_add_kernel<<<scatterGrid, blk, 0, stream>>>(hB, src, dst, agg);
    gemm_fused_kernel<<<gemmGrid, blk, 0, stream>>>(agg, hB, Wrel3h, Wroot3h,
                                                    b_rel3, hA, 0);

    // ---- Global max pool + MLP head
    fill_u32_kernel<<<64, blk, 0, stream>>>(gmono, 0u, (long)N_GRAPHS * DIM);
    seg_max_kernel<<<segGrid, blk, 0, stream>>>(hA, batch, gmono);
    mlp_kernel<<<1, 64, 0, stream>>>(gmono, W1, b1, W2, b2, out);
}
